// CAWNResonanceBlock_68195490726315
// MI455X (gfx1250) — matrix operand order repaired and checked by reading, not compiled, not run on hardware
//
#include <hip/hip_runtime.h>

// ---------------------------------------------------------------------------
// CAWN Resonance Block for MI455X (gfx1250, wave32, WMMA bf16)
// Memory-bound workload (~86 GFLOP vs ~1 GB traffic @ 23.3 TB/s): bf16
// operands everywhere, deep fusion, WMMA for all three GEMMs, chunked scan.
// GEMM v3: 16x64 C-tile per wave, double-buffered GLOBAL_LOAD_ASYNC_TO_LDS
// pipeline (stage chunk c+1 while computing chunk c), batched B-frag loads.
// ---------------------------------------------------------------------------

#ifndef __has_builtin
#define __has_builtin(x) 0
#endif

#if __has_builtin(__builtin_amdgcn_global_load_async_to_lds_b128)
#define GEMM_ASYNC 1
#else
#define GEMM_ASYNC 0
#endif

typedef __bf16 bf16_t;
typedef __attribute__((ext_vector_type(16))) __bf16 v16bf;
typedef __attribute__((ext_vector_type(8)))  __bf16 v8bf;
typedef __attribute__((ext_vector_type(8)))  float  v8f;

constexpr int B_ = 8, S_ = 4096, D_ = 1024, R_ = 128;
constexpr int M_ = B_ * S_;                 // 32768 rows
constexpr int NCH = 16, CHL = S_ / NCH;     // scan chunks: 16 x 256
constexpr float RMS_EPS = 1.1920929e-07f;   // finfo(float32).eps

// ---- workspace layout (bytes) --------------------------------------------
constexpr size_t OFF_HBF  = 0;                                  // M*D bf16 (h, later h2)
constexpr size_t OFF_SIG  = OFF_HBF  + (size_t)M_ * D_ * 2;     // M*R f32
constexpr size_t OFF_ACC  = OFF_SIG  + (size_t)M_ * R_ * 4;     // M*R bf16
constexpr size_t OFF_INV  = OFF_ACC  + (size_t)M_ * R_ * 2;     // M f32
constexpr size_t OFF_XM   = OFF_INV  + (size_t)M_ * 4;          // B*D f32
constexpr size_t OFF_GATE = OFF_XM   + (size_t)B_ * D_ * 4;     // B*R f32
constexpr size_t OFF_RET  = OFF_GATE + (size_t)B_ * R_ * 4;     // B*R f32
constexpr size_t OFF_FN   = OFF_RET  + (size_t)B_ * R_ * 4;     // R*D bf16
constexpr size_t OFF_WO   = OFF_FN   + (size_t)R_ * D_ * 2;     // D*R bf16
constexpr size_t OFF_WF   = OFF_WO   + (size_t)D_ * R_ * 2;     // D*D bf16
constexpr size_t OFF_CEND = OFF_WF   + (size_t)D_ * D_ * 2;     // B*NCH*R f32
constexpr size_t OFF_CIN  = OFF_CEND + (size_t)B_ * NCH * R_ * 4;
constexpr size_t OFF_XMP  = OFF_CIN  + (size_t)B_ * NCH * R_ * 4; // 16*B*D f32 partials

__device__ __forceinline__ float sigmoidf_(float v) { return 1.0f / (1.0f + __expf(-v)); }
__device__ __forceinline__ float siluf_(float v)    { return v * sigmoidf_(v); }

__device__ __forceinline__ float block_reduce256(float v, float* red) {
    int t = threadIdx.x;
    red[t] = v; __syncthreads();
    #pragma unroll
    for (int o = 128; o > 0; o >>= 1) {
        if (t < o) red[t] += red[t + o];
        __syncthreads();
    }
    float r = red[0]; __syncthreads();
    return r;
}

// ---- async memory -> LDS helpers (CDNA5 GLOBAL_LOAD_ASYNC_TO_LDS_B128) ----
// Builtin param types (from hipcc diagnostic): int4 addrspace(1)* source,
// int4 addrspace(3)* LDS destination. Generic->AS casts via integer round-trip.
__device__ __forceinline__ void stage16(const bf16_t* g, bf16_t* l) {
#if GEMM_ASYNC
    typedef int v4i_raw __attribute__((__vector_size__(4 * sizeof(int))));
    typedef __attribute__((address_space(1))) v4i_raw as1_v4i;
    typedef __attribute__((address_space(3))) v4i_raw as3_v4i;
    __builtin_amdgcn_global_load_async_to_lds_b128(
        (as1_v4i*)(unsigned long long)g,
        (as3_v4i*)(unsigned int)(unsigned long long)l,
        0, 0);
#else
    *reinterpret_cast<v8bf*>(l) = *reinterpret_cast<const v8bf*>(g);
#endif
}

__device__ __forceinline__ void stage_wait() {
#if GEMM_ASYNC
#if __has_builtin(__builtin_amdgcn_s_wait_asynccnt)
    __builtin_amdgcn_s_wait_asynccnt(0);
#else
    asm volatile("s_wait_asynccnt 0x0" ::: "memory");
#endif
#endif
}

// ---- small converters -----------------------------------------------------
__global__ void k_cvt(const float* __restrict__ s, bf16_t* __restrict__ d, int n) {
    int i = blockIdx.x * blockDim.x + threadIdx.x;
    if (i < n) d[i] = (bf16_t)s[i];
}

__global__ void k_fnorm(const float* __restrict__ f, bf16_t* __restrict__ fn) {
    __shared__ float red[256];
    int r = blockIdx.x, t = threadIdx.x;
    const float* fr = f + (size_t)r * D_;
    float v[4]; float ss = 0.f;
    #pragma unroll
    for (int j = 0; j < 4; j++) { v[j] = fr[t + j * 256]; ss += v[j] * v[j]; }
    float ns  = block_reduce256(ss, red);
    float inv = 1.f / fmaxf(sqrtf(ns), 1e-12f);
    #pragma unroll
    for (int j = 0; j < 4; j++) fn[(size_t)r * D_ + t + j * 256] = (bf16_t)(v[j] * inv);
}

// ---- fused rmsnorm -> depthwise causal conv(3) -> clip -> silu ------------
__global__ void k_rmsconv(const float* __restrict__ x,  const float* __restrict__ n1w,
                          const float* __restrict__ gam, const float* __restrict__ bet,
                          const float* __restrict__ cw,  bf16_t* __restrict__ h,
                          float* __restrict__ invn) {
    __shared__ float red[256];
    int row = blockIdx.x;
    int s = row & (S_ - 1);
    int t = threadIdx.x;
    const float* x0 = x + (size_t)row * D_;
    bool p1 = (s >= 1), p2 = (s >= 2);
    float xv0[4], xv1[4], xv2[4];
    float ss0 = 0.f, ss1 = 0.f, ss2 = 0.f;
    #pragma unroll
    for (int j = 0; j < 4; j++) {
        int d = t + j * 256;
        float a = x0[d];                      xv0[j] = a; ss0 += a * a;
        float b = p1 ? x0[d - D_]     : 0.f;  xv1[j] = b; ss1 += b * b;
        float c = p2 ? x0[d - 2 * D_] : 0.f;  xv2[j] = c; ss2 += c * c;
    }
    float m0 = block_reduce256(ss0, red) * (1.f / D_);
    float m1 = block_reduce256(ss1, red) * (1.f / D_);
    float m2 = block_reduce256(ss2, red) * (1.f / D_);
    float sc0 = rsqrtf(m0 + RMS_EPS), sc1 = rsqrtf(m1 + RMS_EPS), sc2 = rsqrtf(m2 + RMS_EPS);

    float hv[4]; float ssh = 0.f;
    #pragma unroll
    for (int j = 0; j < 4; j++) {
        int d = t + j * 256;
        float w = n1w[d] * gam[d], bb = bet[d];
        float q0 = xv0[j] * sc0 * w + bb;
        float q1 = p1 ? (xv1[j] * sc1 * w + bb) : 0.f;
        float q2 = p2 ? (xv2[j] * sc2 * w + bb) : 0.f;
        float c  = cw[d * 3 + 0] * q2 + cw[d * 3 + 1] * q1 + cw[d * 3 + 2] * q0;
        c = fminf(fmaxf(c, -50.f), 50.f);
        float sv = siluf_(c);
        hv[j] = sv; ssh += sv * sv;
    }
    float ns = block_reduce256(ssh, red);
    if (t == 0) invn[row] = 1.f / fmaxf(sqrtf(ns), 1e-12f);
    bf16_t* hr = h + (size_t)row * D_;
    #pragma unroll
    for (int j = 0; j < 4; j++) hr[t + j * 256] = (bf16_t)hv[j];
}

// ---- deterministic column mean of h over S (two stage, no atomics) --------
__global__ void k_xmean_part(const bf16_t* __restrict__ h, float* __restrict__ part) {
    int g = blockIdx.x;
    int sc = g & 15; g >>= 4;
    int dblk = g & 3; int b = g >> 2;
    int d = dblk * 256 + threadIdx.x;
    const bf16_t* p = h + ((size_t)(b * S_ + sc * 256)) * D_ + d;
    float sum = 0.f;
    for (int i = 0; i < 256; i++) sum += (float)p[(size_t)i * D_];
    part[(size_t)sc * (B_ * D_) + b * D_ + d] = sum;
}

__global__ void k_xmean_sum(const float* __restrict__ part, float* __restrict__ xm) {
    int i = blockIdx.x * blockDim.x + threadIdx.x;   // over B*D
    float s = 0.f;
    for (int sc = 0; sc < 16; sc++) s += part[(size_t)sc * (B_ * D_) + i];
    xm[i] = s;
}

// ---- gate / retention sigmoids --------------------------------------------
__global__ void k_gates(const float* __restrict__ xm, const float* __restrict__ wg,
                        const float* __restrict__ wr, const float* __restrict__ gb,
                        const float* __restrict__ rb, float* __restrict__ gate,
                        float* __restrict__ ret) {
    __shared__ float red[256];
    int idx = blockIdx.x; int b = idx >> 7, r = idx & 127;
    int t = threadIdx.x;
    const float* xb  = xm + (size_t)b * D_;
    const float* wgr = wg + (size_t)r * D_;
    const float* wrr = wr + (size_t)r * D_;
    float dg = 0.f, dr = 0.f;
    #pragma unroll
    for (int j = 0; j < 4; j++) {
        int d = t + j * 256;
        float v = xb[d] * (1.f / S_);
        dg += v * wgr[d];
        dr += v * wrr[d];
    }
    float sg = block_reduce256(dg, red);
    float sr = block_reduce256(dr, red);
    if (t == 0) {
        float gs = sigmoidf_(sg + gb[r]);
        gate[b * R_ + r] = (gs >= 0.001f) ? gs : 0.f;   // straight-through forward
        ret[b * R_ + r]  = sigmoidf_(sr + rb[r]);
    }
}

// ---- WMMA bf16 GEMM v3: C = A(MxK) . W(NxK)^T ------------------------------
// Block = 8 waves. Each wave: 16x64 C tile (4 accumulators), A-frag reused
// across 4 WMMAs. Double-buffered async stage: chunk c+1 streams into the
// alternate LDS buffer while chunk c is consumed; asynccnt wait + barrier
// only at chunk boundaries. B-frags batch-preloaded (1 clause, 1 wait, 4 WMMA).
// NTOT=128 : 2 N-waves x 4 M-waves -> block tile 64x128, grid (M/64, 1)
// NTOT=1024: 4 N-waves x 2 M-waves -> block tile 32x256, grid (M/32, 4)
// EPI 0: sig = c*invnorm*gate   EPI 1: out = x + c   EPI 2: out = x2 + silu(c)
template <int KDIM, int NTOT, int EPI>
__global__ void k_gemm(const bf16_t* __restrict__ A, const bf16_t* __restrict__ W,
                       const float* __restrict__ aux0, const float* __restrict__ aux1,
                       float* __restrict__ out) {
    constexpr int WN  = (NTOT == 128) ? 2 : 4;   // waves along N
    constexpr int WM  = 8 / WN;                  // waves along M
    constexpr int RB  = WM * 16;                 // A rows per block
    constexpr int KC  = 64;                      // K chunk per LDS stage
    constexpr int LST = KC + 8;                  // padded LDS row stride (bf16)
    constexpr int CHUNKS = RB * (KC / 8);        // 16B chunks per stage
    constexpr int ITER   = CHUNKS / 256;         // staging loads per thread
    constexpr int NCHUNK = KDIM / KC;

    __shared__ __align__(16) bf16_t At[2][RB * LST];
    int tid  = threadIdx.x;
    int wave = tid >> 5, lane = tid & 31;
    int half = lane >> 4, lrow = lane & 15;
    int wm = wave / WN, wn = wave % WN;
    int m0    = blockIdx.x * RB;
    int nbase = blockIdx.y * (WN * 64) + wn * 64;

    union FA { v16bf v; v8bf h[2]; };
    v8f acc[4] = {};
    const bf16_t* Wr0 = W + (size_t)(nbase + lrow) * KDIM;
    int ar = wm * 16 + lrow;                     // this lane's A row in LDS

    auto stage_chunk = [&](int k0, int buf) {
        #pragma unroll
        for (int it = 0; it < ITER; it++) {      // all 256 lanes, full EXEC
            int idx = tid + it * 256;
            int rr = idx >> 3, ch = idx & 7;
            stage16(&A[(size_t)(m0 + rr) * KDIM + k0 + ch * 8],
                    &At[buf][rr * LST + ch * 8]);
        }
    };

    stage_chunk(0, 0);                           // prologue: fill buffer 0
    stage_wait();
    __syncthreads();

    for (int c = 0; c < NCHUNK; c++) {
        int cur = c & 1;
        if (c + 1 < NCHUNK) stage_chunk((c + 1) * KC, cur ^ 1);  // overlap DMA
        int k0 = c * KC;
        #pragma unroll
        for (int kk = 0; kk < KC; kk += 32) {
            FA a;
            a.h[0] = *reinterpret_cast<const v8bf*>(&At[cur][ar * LST + kk + half * 8]);
            a.h[1] = *reinterpret_cast<const v8bf*>(&At[cur][ar * LST + kk + 16 + half * 8]);
            FA b[4];
            #pragma unroll
            for (int j = 0; j < 4; j++) {        // batched B-frag clause
                const bf16_t* Wr = Wr0 + (size_t)j * 16 * KDIM;
                b[j].h[0] = *reinterpret_cast<const v8bf*>(&Wr[k0 + kk + half * 8]);
                b[j].h[1] = *reinterpret_cast<const v8bf*>(&Wr[k0 + kk + 16 + half * 8]);
            }
            #pragma unroll
            for (int j = 0; j < 4; j++) {        // back-to-back WMMAs
                acc[j] = __builtin_amdgcn_wmma_f32_16x16x32_bf16(
                             false, a.v, false, b[j].v, (short)0, acc[j], false, false);
            }
        }
        stage_wait();                            // next buffer's DMA complete
        __syncthreads();                         // everyone done reading cur
    }

    int row0 = m0 + wm * 16;
    #pragma unroll
    for (int j = 0; j < 4; j++) {
        int n = nbase + j * 16 + lrow;
        #pragma unroll
        for (int i = 0; i < 8; i++) {
            int row = row0 + i + 8 * half;       // C layout: VGPR i -> M=i+8*half
            float c = acc[j][i];
            if (EPI == 0) {
                out[(size_t)row * R_ + n] = c * aux0[row] * aux1[(row >> 12) * R_ + n];
            } else if (EPI == 1) {
                size_t idx = (size_t)row * D_ + n;
                out[idx] = aux0[idx] + c;
            } else {
                size_t idx = (size_t)row * D_ + n;
                out[idx] = aux0[idx] + siluf_(c);
            }
        }
    }
}

// ---- chunked linear recurrence: state = sig + ret*state --------------------
__global__ void k_scan_end(const float* __restrict__ sig, const float* __restrict__ ret,
                           float* __restrict__ cend) {
    int g = blockIdx.x; int b = g >> 4, c = g & 15;
    int r = threadIdx.x;
    float rt = ret[b * R_ + r];
    const float* p = sig + ((size_t)(b * S_ + c * CHL)) * R_ + r;
    float y = 0.f;
    for (int t = 0; t < CHL; t++) y = p[(size_t)t * R_] + y * rt;
    cend[(b * NCH + c) * R_ + r] = y;
}

__global__ void k_scan_combine(const float* __restrict__ cend, const float* __restrict__ ret,
                               const float* __restrict__ prev, float* __restrict__ cin,
                               float* __restrict__ nstate) {
    int b = blockIdx.x; int r = threadIdx.x;
    float rt = ret[b * R_ + r];
    float rl = __powf(rt, (float)CHL);           // ret^256
    float st = prev[b * R_ + r];
    for (int c = 0; c < NCH; c++) {
        cin[(b * NCH + c) * R_ + r] = st;
        st = cend[(b * NCH + c) * R_ + r] + rl * st;
    }
    nstate[b * R_ + r] = st;                     // final resonance state output
}

__global__ void k_scan_final(const float* __restrict__ sig, const float* __restrict__ ret,
                             const float* __restrict__ cin, bf16_t* __restrict__ accb) {
    int g = blockIdx.x; int b = g >> 4, c = g & 15;
    int r = threadIdx.x;
    float rt = ret[b * R_ + r];
    float y  = cin[(b * NCH + c) * R_ + r];
    const float* p = sig  + ((size_t)(b * S_ + c * CHL)) * R_ + r;
    bf16_t*      q = accb + ((size_t)(b * S_ + c * CHL)) * R_ + r;
    for (int t = 0; t < CHL; t++) {
        y = p[(size_t)t * R_] + rt * y;
        q[(size_t)t * R_] = (bf16_t)y;
    }
}

// ---- second rmsnorm -> bf16 ------------------------------------------------
__global__ void k_h2(const float* __restrict__ x2, const float* __restrict__ n2w,
                     const float* __restrict__ gam, const float* __restrict__ bet,
                     bf16_t* __restrict__ h2) {
    __shared__ float red[256];
    int row = blockIdx.x, t = threadIdx.x;
    const float* xr = x2 + (size_t)row * D_;
    float v[4]; float ss = 0.f;
    #pragma unroll
    for (int j = 0; j < 4; j++) { v[j] = xr[t + j * 256]; ss += v[j] * v[j]; }
    float m  = block_reduce256(ss, red) * (1.f / D_);
    float sc = rsqrtf(m + RMS_EPS);
    #pragma unroll
    for (int j = 0; j < 4; j++) {
        int d = t + j * 256;
        h2[(size_t)row * D_ + d] = (bf16_t)(v[j] * sc * n2w[d] * gam[d] + bet[d]);
    }
}

// ---------------------------------------------------------------------------
extern "C" void kernel_launch(void* const* d_in, const int* in_sizes, int n_in,
                              void* d_out, int out_size, void* d_ws, size_t ws_size,
                              hipStream_t stream) {
    const float* x     = (const float*)d_in[0];
    const float* gam   = (const float*)d_in[1];
    const float* bet   = (const float*)d_in[2];
    const float* prev  = (const float*)d_in[3];
    const float* cw    = (const float*)d_in[4];
    const float* freq  = (const float*)d_in[5];
    const float* rb    = (const float*)d_in[6];
    const float* wret  = (const float*)d_in[7];
    const float* wgate = (const float*)d_in[8];
    const float* gb    = (const float*)d_in[9];
    const float* wout  = (const float*)d_in[10];
    const float* wffn  = (const float*)d_in[11];
    const float* n1w   = (const float*)d_in[12];
    const float* n2w   = (const float*)d_in[13];
    float* out = (float*)d_out;                  // [M*D floats] + [B*R new_state]

    char* ws = (char*)d_ws;
    bf16_t* hbf  = (bf16_t*)(ws + OFF_HBF);
    float*  sig  = (float*) (ws + OFF_SIG);
    bf16_t* accb = (bf16_t*)(ws + OFF_ACC);
    float*  invn = (float*) (ws + OFF_INV);
    float*  xm   = (float*) (ws + OFF_XM);
    float*  gate = (float*) (ws + OFF_GATE);
    float*  ret  = (float*) (ws + OFF_RET);
    bf16_t* fnb  = (bf16_t*)(ws + OFF_FN);
    bf16_t* wob  = (bf16_t*)(ws + OFF_WO);
    bf16_t* wfb  = (bf16_t*)(ws + OFF_WF);
    float*  cend = (float*) (ws + OFF_CEND);
    float*  cin  = (float*) (ws + OFF_CIN);
    float*  xmp  = (float*) (ws + OFF_XMP);

    // weights -> bf16, normalized frequencies -> bf16
    k_cvt<<<(D_ * R_ + 255) / 256, 256, 0, stream>>>(wout, wob, D_ * R_);
    k_cvt<<<(D_ * D_ + 255) / 256, 256, 0, stream>>>(wffn, wfb, D_ * D_);
    k_fnorm<<<R_, 256, 0, stream>>>(freq, fnb);

    // fused rmsnorm + conv + silu -> h (bf16) + 1/||h|| per row
    k_rmsconv<<<M_, 256, 0, stream>>>(x, n1w, gam, bet, cw, hbf, invn);

    // deterministic x_mean over S
    k_xmean_part<<<B_ * 4 * 16, 256, 0, stream>>>(hbf, xmp);
    k_xmean_sum<<<(B_ * D_) / 256, 256, 0, stream>>>(xmp, xm);
    k_gates<<<B_ * R_, 256, 0, stream>>>(xm, wgate, wret, gb, rb, gate, ret);

    // resonance GEMM (WMMA bf16), epilogue: * invnorm * gate -> sig
    k_gemm<1024, 128, 0><<<dim3(M_ / 64, 1), 256, 0, stream>>>(hbf, fnb, invn, gate, sig);

    // chunked scan; combine writes new_state into d_out tail
    k_scan_end<<<B_ * NCH, R_, 0, stream>>>(sig, ret, cend);
    k_scan_combine<<<B_, R_, 0, stream>>>(cend, ret, prev, cin, out + (size_t)M_ * D_);
    k_scan_final<<<B_ * NCH, R_, 0, stream>>>(sig, ret, cin, accb);

    // x2 = x + acc @ w_out^T  (WMMA), stored in d_out
    k_gemm<128, 1024, 1><<<dim3(M_ / 32, 4), 256, 0, stream>>>(accb, wob, x, nullptr, out);

    // h2 = rmsnorm(x2)*gamma+beta (bf16, reuses h buffer)
    k_h2<<<M_, 256, 0, stream>>>(out, n2w, gam, bet, hbf);

    // out = x2 + silu(h2 @ w_ffn^T)  (WMMA), RMW on d_out
    k_gemm<1024, 1024, 2><<<dim3(M_ / 32, 4), 256, 0, stream>>>(hbf, wfb, out, nullptr, out);
}
